// GINJK_88244398063736
// MI455X (gfx1250) — matrix operand order repaired and checked by reading, not compiled, not run on hardware
//
#include <hip/hip_runtime.h>
#include <hip/hip_bf16.h>
#include <stdint.h>

// Problem constants (match reference)
#define NN   100000
#define EE   1600000
#define INC  128
#define HC   64
#define GG   64
#define OUTC 16

typedef __attribute__((ext_vector_type(16))) __bf16 v16bf;
typedef __attribute__((ext_vector_type(8)))  float  v8f;

union AFrag {
    v16bf v;
    __bf16 h[16];
    uint4  u[2];
};

// ---------------------------------------------------------------------------
// Weight -> bf16 B-fragment layout converter.
// W is [K,64] row-major f32. Output fragment layout, per 32x16 (KxN) tile:
//   frag fi = kb*4 + nb ; element e = fi*512 + lane*16 + s  (s = 2r+hi)
//   lane<16 : krow = kb*32 +     (s&7) + 16*(s>>3), col = nb*16 + lane
//   lane>=16: krow = kb*32 + 8 + (s&7) + 16*(s>>3), col = nb*16 + lane-16
// This matches V_WMMA_*_16X16X32 B-matrix VGPR striping (ISA 7.12.2).
// ---------------------------------------------------------------------------
__global__ void wfrag_conv_kernel(const float* __restrict__ W,
                                  __bf16* __restrict__ out, int K) {
    int e = blockIdx.x * blockDim.x + threadIdx.x;
    int total = K * 64;
    if (e >= total) return;
    int fi   = e >> 9;
    int rem  = e & 511;
    int lane = rem >> 4;
    int s    = rem & 15;
    int kb = fi >> 2;
    int nb = fi & 3;
    int r = s >> 1, hi = s & 1;
    int krow = kb * 32 + ((lane < 16) ? 0 : 8) + (s & 7) + ((s >> 3) << 4);
    int col  = nb * 16 + (lane & 15);
    out[e] = (__bf16)W[krow * 64 + col];
}

// ---------------------------------------------------------------------------
// Vectorized copy (agg init: agg = x  /  agg2 = h1), covers the (1+eps)*x term
// ---------------------------------------------------------------------------
__global__ void copy_f4_kernel(float4* __restrict__ dst,
                               const float4* __restrict__ src, int n4) {
    int i = blockIdx.x * blockDim.x + threadIdx.x;
    if (i < n4) dst[i] = src[i];
}

// ---------------------------------------------------------------------------
// Edge scatter-add: agg[dst[e], :] += x[src[e], :].  One thread = 4 features.
// F4 = F/4 (power of two), total = E * F4 threads.
// ---------------------------------------------------------------------------
__global__ __launch_bounds__(256)
void scatter_add_kernel(const float* __restrict__ xin,
                        const int* __restrict__ src,
                        const int* __restrict__ dst,
                        float* __restrict__ agg,
                        int log2F4, long long total) {
    long long idx = (long long)blockIdx.x * blockDim.x + threadIdx.x;
    if (idx >= total) return;
    const int F4 = 1 << log2F4;
    long long e = idx >> log2F4;
    int c = (int)(idx & (F4 - 1));
    int s = src[e];
    int d = dst[e];
    float4 xv = ((const float4*)xin)[(size_t)s * F4 + c];
    float* ap = agg + ((size_t)d * F4 + c) * 4;
    atomicAdd(ap + 0, xv.x);
    atomicAdd(ap + 1, xv.y);
    atomicAdd(ap + 2, xv.z);
    atomicAdd(ap + 3, xv.w);
}

// ---------------------------------------------------------------------------
// Fused GIN MLP:  h = relu( relu(agg @ W1 + b1) @ W2 + b2 )
// One wave computes a 16-row tile. GEMM1: A built from f32 activations
// (ISA 16-bit A 16x32 layout: lane holds row (lane&15), two 8-element K runs).
// Intermediate t is bounced through wave-private LDS (bf16) to re-fragment
// as A for GEMM2. All matmuls: v_wmma_f32_16x16x32_bf16, f32 accumulate.
// ---------------------------------------------------------------------------
template <int K1>
__global__ __launch_bounds__(256)
void gin_mlp_kernel(const float* __restrict__ agg,
                    const __bf16* __restrict__ wf1,
                    const float*  __restrict__ b1,
                    const __bf16* __restrict__ wf2,
                    const float*  __restrict__ b2,
                    float* __restrict__ hout) {
    __shared__ __bf16 tbuf[8 * 16 * 64];   // 16KB: one 16x64 bf16 tile per wave

    const int lane = threadIdx.x & 31;
    const int wid  = threadIdx.x >> 5;
    const int tile = blockIdx.x * 8 + wid;
    if (tile >= NN / 16) return;           // wave-uniform exit; EXEC stays full

    const int rl   = lane & 15;
    const int koff = (lane < 16) ? 0 : 8;
    const size_t row = (size_t)tile * 16 + rl;

    // ---- GEMM1: [16,K1] x [K1,64] ----
    v8f acc[4] = {};
    #pragma unroll
    for (int kb = 0; kb < K1 / 32; ++kb) {
        AFrag a;
        const float* p = agg + row * K1 + kb * 32 + koff;
        float4 q0 = *(const float4*)(p);
        float4 q1 = *(const float4*)(p + 4);
        float4 q2 = *(const float4*)(p + 16);
        float4 q3 = *(const float4*)(p + 20);
        a.h[0] = (__bf16)q0.x; a.h[1] = (__bf16)q0.y;
        a.h[2] = (__bf16)q0.z; a.h[3] = (__bf16)q0.w;
        a.h[4] = (__bf16)q1.x; a.h[5] = (__bf16)q1.y;
        a.h[6] = (__bf16)q1.z; a.h[7] = (__bf16)q1.w;
        a.h[8]  = (__bf16)q2.x; a.h[9]  = (__bf16)q2.y;
        a.h[10] = (__bf16)q2.z; a.h[11] = (__bf16)q2.w;
        a.h[12] = (__bf16)q3.x; a.h[13] = (__bf16)q3.y;
        a.h[14] = (__bf16)q3.z; a.h[15] = (__bf16)q3.w;
        #pragma unroll
        for (int nb = 0; nb < 4; ++nb) {
            v16bf b = *(const v16bf*)(wf1 + (size_t)(kb * 4 + nb) * 512 + lane * 16);
            acc[nb] = __builtin_amdgcn_wmma_f32_16x16x32_bf16(
                false, a.v, false, b, (short)0, acc[nb], false, false);
        }
    }

    // ---- bias + relu -> wave-private LDS tile (bf16, row-major 16x64) ----
    __bf16* tb = tbuf + wid * 1024;
    #pragma unroll
    for (int nb = 0; nb < 4; ++nb) {
        float bias = b1[nb * 16 + rl];
        #pragma unroll
        for (int v = 0; v < 8; ++v) {
            int m = v + ((lane < 16) ? 0 : 8);     // C/D layout (ISA 7.12.2)
            float val = acc[nb][v] + bias;
            val = val > 0.f ? val : 0.f;
            tb[m * 64 + nb * 16 + rl] = (__bf16)val;
        }
    }
    asm volatile("s_wait_dscnt 0" ::: "memory");   // same-wave LDS RAW

    // ---- GEMM2: [16,64] x [64,64] ----
    v8f acc2[4] = {};
    #pragma unroll
    for (int kb = 0; kb < 2; ++kb) {
        AFrag a;
        const __bf16* p = tb + rl * 64 + kb * 32 + koff;
        a.u[0] = *(const uint4*)(p);        // 8 bf16 = 16B aligned
        a.u[1] = *(const uint4*)(p + 16);
        #pragma unroll
        for (int nb = 0; nb < 4; ++nb) {
            v16bf b = *(const v16bf*)(wf2 + (size_t)(kb * 4 + nb) * 512 + lane * 16);
            acc2[nb] = __builtin_amdgcn_wmma_f32_16x16x32_bf16(
                false, a.v, false, b, (short)0, acc2[nb], false, false);
        }
    }

    // ---- bias + relu -> global h (f32) ----
    #pragma unroll
    for (int nb = 0; nb < 4; ++nb) {
        float bias = b2[nb * 16 + rl];
        #pragma unroll
        for (int v = 0; v < 8; ++v) {
            int m = v + ((lane < 16) ? 0 : 8);
            float val = acc2[nb][v] + bias;
            val = val > 0.f ? val : 0.f;
            hout[((size_t)tile * 16 + m) * 64 + nb * 16 + rl] = val;
        }
    }
}

// ---------------------------------------------------------------------------
// Global add-pool of [h1|h2] by graph id, via per-block 64x128 LDS accumulator
// (32KB of the 320KB WGP LDS) -> ~8x fewer global atomics.
// ---------------------------------------------------------------------------
__global__ __launch_bounds__(256)
void pool_kernel(const float* __restrict__ h1, const float* __restrict__ h2,
                 const int* __restrict__ batch, float* __restrict__ pooled,
                 int nNodes) {
    __shared__ float sm[GG * 128];
    for (int i = threadIdx.x; i < GG * 128; i += 256) sm[i] = 0.f;
    __syncthreads();

    const int f    = threadIdx.x & 127;
    const int nsub = threadIdx.x >> 7;     // 0..1
    const int base = blockIdx.x * 512;
    int stop = base + 512;
    if (stop > nNodes) stop = nNodes;
    for (int n = base + nsub; n < stop; n += 2) {
        int g = batch[n];
        float val = (f < 64) ? h1[(size_t)n * 64 + f]
                             : h2[(size_t)n * 64 + (f - 64)];
        atomicAdd(&sm[g * 128 + f], val);  // ds_add_f32, bank-conflict free
    }
    __syncthreads();
    for (int i = threadIdx.x; i < GG * 128; i += 256)
        atomicAdd(&pooled[i], sm[i]);
}

// ---------------------------------------------------------------------------
// Head: logits = pooled @ Wfc + bfc ; log_softmax per row.
// 16-lane shuffle reductions (wave32: two rows of 16 per wave).
// ---------------------------------------------------------------------------
__global__ __launch_bounds__(256)
void head_kernel(const float* __restrict__ pooled,
                 const float* __restrict__ Wfc,
                 const float* __restrict__ bfc,
                 float* __restrict__ out) {
    int idx = blockIdx.x * blockDim.x + threadIdx.x;
    if (idx >= GG * OUTC) return;
    int g = idx >> 4, o = idx & 15;
    float acc = bfc[o];
    #pragma unroll 4
    for (int k = 0; k < 2 * HC; ++k)
        acc += pooled[g * 128 + k] * Wfc[k * OUTC + o];
    float m = acc;
    #pragma unroll
    for (int off = 8; off; off >>= 1) {
        float t = __shfl_xor(m, off, 16);
        m = t > m ? t : m;
    }
    float s = __expf(acc - m);
    #pragma unroll
    for (int off = 8; off; off >>= 1) s += __shfl_xor(s, off, 16);
    out[idx] = acc - m - __logf(s);
}

// ---------------------------------------------------------------------------
extern "C" void kernel_launch(void* const* d_in, const int* in_sizes, int n_in,
                              void* d_out, int out_size, void* d_ws, size_t ws_size,
                              hipStream_t stream) {
    const float* x    = (const float*)d_in[0];
    const int*   ei   = (const int*)d_in[1];
    const int*   batch= (const int*)d_in[2];
    const float* W1a  = (const float*)d_in[3];
    const float* b1a  = (const float*)d_in[4];
    const float* W2a  = (const float*)d_in[5];
    const float* b2a  = (const float*)d_in[6];
    const float* W1b  = (const float*)d_in[7];
    const float* b1b  = (const float*)d_in[8];
    const float* W2b  = (const float*)d_in[9];
    const float* b2b  = (const float*)d_in[10];
    const float* Wfc  = (const float*)d_in[11];
    const float* bfc  = (const float*)d_in[12];

    const int* src = ei;
    const int* dst = ei + EE;

    // Workspace layout (all 256B aligned)
    char* ws = (char*)d_ws;
    float* agg    = (float*)(ws);                                // N*128 f32 (reused)
    float* h1     = (float*)(ws + (size_t)NN * 128 * 4);         // N*64
    float* h2     = (float*)(ws + (size_t)NN * 128 * 4 + (size_t)NN * 64 * 4);
    float* pooled = (float*)(ws + (size_t)NN * 128 * 4 + (size_t)NN * 64 * 8);
    __bf16* wf1a  = (__bf16*)((char*)pooled + GG * 128 * 4);
    __bf16* wf2a  = wf1a + 128 * 64;
    __bf16* wf1b  = wf2a + 64 * 64;
    __bf16* wf2b  = wf1b + 64 * 64;

    // 0) pre-swizzle weights to bf16 WMMA B-fragment layout
    wfrag_conv_kernel<<<(128 * 64 + 255) / 256, 256, 0, stream>>>(W1a, wf1a, 128);
    wfrag_conv_kernel<<<(64 * 64 + 255) / 256, 256, 0, stream>>>(W2a, wf2a, 64);
    wfrag_conv_kernel<<<(64 * 64 + 255) / 256, 256, 0, stream>>>(W1b, wf1b, 64);
    wfrag_conv_kernel<<<(64 * 64 + 255) / 256, 256, 0, stream>>>(W2b, wf2b, 64);

    // 1) layer 1: agg = x + scatter(x)
    copy_f4_kernel<<<(NN * 128 / 4 + 255) / 256, 256, 0, stream>>>(
        (float4*)agg, (const float4*)x, NN * 128 / 4);
    scatter_add_kernel<<<(int)(((long long)EE * 32 + 255) / 256), 256, 0, stream>>>(
        x, src, dst, agg, 5, (long long)EE * 32);
    // h1 = relu(MLP1(agg))
    gin_mlp_kernel<128><<<(NN / 16 + 7) / 8, 256, 0, stream>>>(
        agg, wf1a, b1a, wf2a, b2a, h1);

    // 2) layer 2: agg2 = h1 + scatter(h1)   (reuses agg buffer, now [N,64])
    copy_f4_kernel<<<(NN * 64 / 4 + 255) / 256, 256, 0, stream>>>(
        (float4*)agg, (const float4*)h1, NN * 64 / 4);
    scatter_add_kernel<<<(int)(((long long)EE * 16 + 255) / 256), 256, 0, stream>>>(
        h1, src, dst, agg, 4, (long long)EE * 16);
    gin_mlp_kernel<64><<<(NN / 16 + 7) / 8, 256, 0, stream>>>(
        agg, wf1b, b1b, wf2b, b2b, h2);

    // 3) pool [h1|h2] by graph
    hipMemsetAsync(pooled, 0, GG * 128 * 4, stream);
    pool_kernel<<<(NN + 511) / 512, 256, 0, stream>>>(h1, h2, batch, pooled, NN);

    // 4) head + log_softmax
    head_kernel<<<(GG * OUTC + 255) / 256, 256, 0, stream>>>(
        pooled, Wfc, bfc, (float*)d_out);
}
